// LinearAttention_63814624084187
// MI455X (gfx1250) — matrix operand order repaired and checked by reading, not compiled, not run on hardware
//
#include <hip/hip_runtime.h>
#include <hip/hip_bf16.h>

// ---------------------------------------------------------------------------
// LinearAttention fused pipeline for MI455X (gfx1250, wave32, WMMA bf16)
//   B=16, C=128, HEADS=4, DH=32, INNER=128, H=W=128, N=16384
// ---------------------------------------------------------------------------

#define B_SZ   16
#define C_CH   128
#define HEADS  4
#define DH     32
#define INNER  128
#define HIMG   128
#define WIMG   128
#define N_TOK  16384
#define SQRT_DH 5.6568542494923806f
#define LN_EPS  1e-5f

// workspace layout (bytes)
#define QKV_ELEMS   ((size_t)B_SZ * 384 * N_TOK)          // bf16
#define STATS_OFF   (QKV_ELEMS * 2)                       // 2048*2 floats
#define CTX_OFF     (STATS_OFF + (size_t)2048 * 2 * 4)    // 64*32*32 floats

typedef __attribute__((ext_vector_type(16))) __bf16       v16bf;
typedef __attribute__((ext_vector_type(8)))  float        v8f;
typedef __attribute__((ext_vector_type(4)))  unsigned int u32x4;

union FragBF { v16bf v; u32x4 q[2]; };
union PackD  { u32x4 q; unsigned short h[8]; };

__device__ __forceinline__ unsigned short f2bf(float f) {
    unsigned int u = __float_as_uint(f);
    return (unsigned short)((u + 0x7FFFu + ((u >> 16) & 1u)) >> 16);   // RNE
}
__device__ __forceinline__ float bf2f(unsigned int h) {
    return __uint_as_float((h & 0xFFFFu) << 16);
}

__device__ __forceinline__ v8f wmma_bf16(const FragBF& a, const FragBF& b, v8f c) {
    return __builtin_amdgcn_wmma_f32_16x16x32_bf16(false, a.v, false, b.v,
                                                   (short)0, c, false, false);
}

// ---------------------------------------------------------------------------
// Kernel 1: qkv[b, o, n] = sum_c w_qkv[o,c] * x[b,c,n]   (output bf16)
// grid (N/64, 384/64, B), block 256 (8 waves)
// ---------------------------------------------------------------------------
__global__ __launch_bounds__(256) void qkv_gemm(const float* __restrict__ x,
                                                const float* __restrict__ wqkv,
                                                unsigned short* __restrict__ qkv) {
    __shared__ __align__(16) unsigned short Abf[64 * 136];   // [o][c] row-major
    __shared__ __align__(16) unsigned short Bt [64 * 136];   // [n][c] transposed

    const int tid = threadIdx.x;
    const int b  = blockIdx.z;
    const int o0 = blockIdx.y * 64;
    const int n0 = blockIdx.x * 64;

#pragma unroll
    for (int i = 0; i < 8; ++i) {                 // A tile: 64x128 f32 -> bf16
        int lin = tid + i * 256;                  // float4 units
        int row = lin >> 5, c4 = lin & 31;
        float4 v4 = *(const float4*)(wqkv + (size_t)(o0 + row) * C_CH + c4 * 4);
        int base = row * 136 + c4 * 4;
        Abf[base + 0] = f2bf(v4.x); Abf[base + 1] = f2bf(v4.y);
        Abf[base + 2] = f2bf(v4.z); Abf[base + 3] = f2bf(v4.w);
    }
#pragma unroll
    for (int i = 0; i < 8; ++i) {                 // x tile: 128x64 -> Bt[n][c]
        int lin = tid + i * 256;
        int c = lin >> 4, j4 = lin & 15;
        float4 v4 = *(const float4*)(x + ((size_t)b * C_CH + c) * N_TOK + n0 + j4 * 4);
        Bt[(j4 * 4 + 0) * 136 + c] = f2bf(v4.x);
        Bt[(j4 * 4 + 1) * 136 + c] = f2bf(v4.y);
        Bt[(j4 * 4 + 2) * 136 + c] = f2bf(v4.z);
        Bt[(j4 * 4 + 3) * 136 + c] = f2bf(v4.w);
    }
    __syncthreads();

    const int wid = tid >> 5, lane = tid & 31;
    const int r = lane & 15, hi = lane >> 4;
    const int mt  = wid & 3;                      // 0..3  (o sub-tile)
    const int ntb = (wid >> 2) * 2;               // 0 or 2 (n sub-tiles)

    FragBF afr[4];
#pragma unroll
    for (int ks = 0; ks < 4; ++ks) {              // hoist A fragments
        const unsigned short* pa = &Abf[(mt * 16 + r) * 136 + ks * 32 + hi * 8];
        afr[ks].q[0] = *(const u32x4*)(pa);
        afr[ks].q[1] = *(const u32x4*)(pa + 16);
    }
#pragma unroll
    for (int t = 0; t < 2; ++t) {
        int nt = ntb + t;
        v8f acc = {};
#pragma unroll
        for (int ks = 0; ks < 4; ++ks) {
            FragBF bf;
            const unsigned short* pb = &Bt[(nt * 16 + r) * 136 + ks * 32 + hi * 8];
            bf.q[0] = *(const u32x4*)(pb);
            bf.q[1] = *(const u32x4*)(pb + 16);
            acc = wmma_bf16(afr[ks], bf, acc);
        }
#pragma unroll
        for (int i = 0; i < 8; ++i) {
            int o = o0 + mt * 16 + hi * 8 + i;
            int n = n0 + nt * 16 + r;
            qkv[((size_t)b * 384 + o) * N_TOK + n] = f2bf(acc[i]);
        }
    }
}

// ---------------------------------------------------------------------------
// Kernel 2a: per k-row (b,h,d) max and sum(exp) over N.  grid 2048, block 256
// ---------------------------------------------------------------------------
__global__ __launch_bounds__(256) void kstats(const unsigned short* __restrict__ qkv,
                                              float* __restrict__ stats) {
    __shared__ float red[256];
    const int bid = blockIdx.x;            // b*128 + (h*32+d)
    const int b = bid >> 7, rem = bid & 127;
    const unsigned int* row =
        (const unsigned int*)(qkv + ((size_t)b * 384 + 128 + rem) * N_TOK);
    const int tid = threadIdx.x;

    float m = -3.4e38f;
#pragma unroll 4
    for (int i = 0; i < 32; ++i) {
        unsigned int u = row[tid + i * 256];
        m = fmaxf(m, fmaxf(bf2f(u), bf2f(u >> 16)));
    }
    red[tid] = m; __syncthreads();
    for (int s = 128; s > 0; s >>= 1) {
        if (tid < s) red[tid] = fmaxf(red[tid], red[tid + s]);
        __syncthreads();
    }
    float rowmax = red[0]; __syncthreads();

    float ssum = 0.f;
#pragma unroll 4
    for (int i = 0; i < 32; ++i) {
        unsigned int u = row[tid + i * 256];
        ssum += __expf(bf2f(u) - rowmax) + __expf(bf2f(u >> 16) - rowmax);
    }
    red[tid] = ssum; __syncthreads();
    for (int s = 128; s > 0; s >>= 1) {
        if (tid < s) red[tid] += red[tid + s];
        __syncthreads();
    }
    if (tid == 0) { stats[bid * 2] = rowmax; stats[bid * 2 + 1] = red[0]; }
}

// ---------------------------------------------------------------------------
// Kernel 2b: ctx[bh][d][e] += sum_n softmax(k)[d,n] * v[e,n]/N   (split-K x8)
// grid (64, 8), block 256
// ---------------------------------------------------------------------------
__global__ __launch_bounds__(256) void context_k(const unsigned short* __restrict__ qkv,
                                                 const float* __restrict__ stats,
                                                 float* __restrict__ ctx) {
    __shared__ float kx [32 * 128];    // [d][j]
    __shared__ float vvT[128 * 36];    // [j][e]
    __shared__ float smax[32], sinv[32];

    const int bh = blockIdx.x;                 // 0..63
    const int b = bh >> 2, h = bh & 3;
    const int nbase = blockIdx.y * 2048;
    const int tid = threadIdx.x;

    if (tid < 32) {
        int srow = b * 128 + h * 32 + tid;
        smax[tid] = stats[srow * 2];
        sinv[tid] = 1.f / stats[srow * 2 + 1];
    }
    __syncthreads();

    const float invN = 1.f / (float)N_TOK;
    const int d = tid >> 3, e0 = (tid & 7) * 4;
    float a0 = 0.f, a1 = 0.f, a2 = 0.f, a3 = 0.f;

    for (int ch = 0; ch < 16; ++ch) {
        const int n0 = nbase + ch * 128;
#pragma unroll
        for (int i = 0; i < 8; ++i) {
            int lin = tid + i * 256;           // u32 units over 32x128
            int dd = lin >> 6, j2 = lin & 63;
            unsigned int uk = *(const unsigned int*)
                (qkv + ((size_t)b * 384 + 128 + h * 32 + dd) * N_TOK + n0 + j2 * 2);
            float mx = smax[dd], iv = sinv[dd];
            kx[dd * 128 + j2 * 2 + 0] = __expf(bf2f(uk)       - mx) * iv;
            kx[dd * 128 + j2 * 2 + 1] = __expf(bf2f(uk >> 16) - mx) * iv;
            unsigned int uv = *(const unsigned int*)
                (qkv + ((size_t)b * 384 + 256 + h * 32 + dd) * N_TOK + n0 + j2 * 2);
            vvT[(j2 * 2 + 0) * 36 + dd] = bf2f(uv)       * invN;
            vvT[(j2 * 2 + 1) * 36 + dd] = bf2f(uv >> 16) * invN;
        }
        __syncthreads();
#pragma unroll 4
        for (int j = 0; j < 128; ++j) {
            float kk = kx[d * 128 + j];
            float4 v4 = *(const float4*)&vvT[j * 36 + e0];
            a0 = fmaf(kk, v4.x, a0); a1 = fmaf(kk, v4.y, a1);
            a2 = fmaf(kk, v4.z, a2); a3 = fmaf(kk, v4.w, a3);
        }
        __syncthreads();
    }
    float* dst = ctx + ((size_t)bh * 32 + d) * 32 + e0;
    atomicAdd(dst + 0, a0); atomicAdd(dst + 1, a1);
    atomicAdd(dst + 2, a2); atomicAdd(dst + 3, a3);
}

// ---------------------------------------------------------------------------
// Kernel 3: fused q-softmax -> ctx apply (WMMA) -> out proj (WMMA) -> bias
//           -> LayerNorm over W -> store.  grid (H, B), block 256
// ---------------------------------------------------------------------------
__global__ __launch_bounds__(256) void attn_out(const unsigned short* __restrict__ qkv,
                                                const float* __restrict__ ctx,
                                                const float* __restrict__ wout,
                                                const float* __restrict__ bout,
                                                const float* __restrict__ gamma,
                                                const float* __restrict__ beta,
                                                float* __restrict__ out) {
    __shared__ __align__(16) char smem[106496];
    unsigned short* qsT = (unsigned short*)(smem);           // [w][i]  128x136 bf16
    unsigned short* wA  = (unsigned short*)(smem + 34816);   // [m][k]  128x136 bf16
    unsigned short* oaT = (unsigned short*)(smem + 69632);   // [w][e]  128x136 bf16
    float* fin  = (float*)(smem);                            // [o][w]  128x132 f32 (aliases qsT/wA)
    float* red  = (float*)(smem + 104448);                   // 256
    float* red2 = (float*)(smem + 105472);                   // 256

    const int tid = threadIdx.x;
    const int b = blockIdx.y, hrow = blockIdx.x;
    const int n0 = hrow * 128;

    // 1a) raw q -> qsT[w][i]  (transposed, bf16 passthrough)
#pragma unroll
    for (int i = 0; i < 32; ++i) {
        int lin = tid + i * 256;                // u32 units over 128x128
        int irow = lin >> 6, j2 = lin & 63;
        unsigned int u = *(const unsigned int*)
            (qkv + ((size_t)b * 384 + irow) * N_TOK + n0 + j2 * 2);
        qsT[(j2 * 2 + 0) * 136 + irow] = (unsigned short)(u & 0xFFFFu);
        qsT[(j2 * 2 + 1) * 136 + irow] = (unsigned short)(u >> 16);
    }
    // 2) block-diagonal ctx^T -> wA[e][d]
    {
        int row = tid >> 1, cb = (tid & 1) * 64;
        int hr = row >> 5;
#pragma unroll 8
        for (int c = 0; c < 64; ++c) {
            int col = cb + c;
            float v = (hr == (col >> 5))
                ? ctx[((size_t)(b * 4 + hr) * 32 + (col & 31)) * 32 + (row & 31)] : 0.f;
            wA[row * 136 + col] = f2bf(v);
        }
    }
    __syncthreads();

    // 1b) softmax over d per (w, head), scale by sqrt(DH), in place in qsT
#pragma unroll
    for (int t = 0; t < 2; ++t) {
        int id = tid + t * 256;
        int w = id >> 2, h = id & 3;
        unsigned short* p = qsT + w * 136 + h * 32;
        float m = -3.4e38f;
        for (int dd = 0; dd < 32; ++dd) m = fmaxf(m, bf2f(p[dd]));
        float s = 0.f;
        for (int dd = 0; dd < 32; ++dd) s += __expf(bf2f(p[dd]) - m);
        float inv = SQRT_DH / s;
        for (int dd = 0; dd < 32; ++dd) p[dd] = f2bf(__expf(bf2f(p[dd]) - m) * inv);
    }
    __syncthreads();

    const int wid = tid >> 5, lane = tid & 31;
    const int r = lane & 15, hi = lane >> 4;
    const int mrow = wid * 16;

    // 3) GEMM1: oa = ctx_diag x q_soft  -> oaT[w][e] (bf16)
    {
        FragBF afr[4];
#pragma unroll
        for (int ks = 0; ks < 4; ++ks) {
            const unsigned short* pa = &wA[(mrow + r) * 136 + ks * 32 + hi * 8];
            afr[ks].q[0] = *(const u32x4*)(pa);
            afr[ks].q[1] = *(const u32x4*)(pa + 16);
        }
#pragma unroll
        for (int nt = 0; nt < 8; ++nt) {
            v8f acc = {};
#pragma unroll
            for (int ks = 0; ks < 4; ++ks) {
                FragBF bf;
                const unsigned short* pb = &qsT[(nt * 16 + r) * 136 + ks * 32 + hi * 8];
                bf.q[0] = *(const u32x4*)(pb);
                bf.q[1] = *(const u32x4*)(pb + 16);
                acc = wmma_bf16(afr[ks], bf, acc);
            }
            PackD st;
#pragma unroll
            for (int i = 0; i < 8; ++i) st.h[i] = f2bf(acc[i]);
            *(u32x4*)&oaT[(nt * 16 + r) * 136 + mrow + hi * 8] = st.q;
        }
    }
    __syncthreads();

    // 4) w_out -> wA (overwrite ctx_diag)
#pragma unroll
    for (int i = 0; i < 16; ++i) {
        int lin = tid + i * 256;                // float4 units over 128x128
        int row = lin >> 5, c4 = lin & 31;
        float4 v4 = *(const float4*)(wout + (size_t)row * INNER + c4 * 4);
        int base = row * 136 + c4 * 4;
        wA[base + 0] = f2bf(v4.x); wA[base + 1] = f2bf(v4.y);
        wA[base + 2] = f2bf(v4.z); wA[base + 3] = f2bf(v4.w);
    }
    __syncthreads();

    // 5) GEMM2: fin = w_out x oa (+ bias) ; accumulate all 8 n-tiles in regs
    v8f acc8[8];
#pragma unroll
    for (int nt = 0; nt < 8; ++nt) acc8[nt] = (v8f){};
    {
        FragBF afr[4];
#pragma unroll
        for (int ks = 0; ks < 4; ++ks) {
            const unsigned short* pa = &wA[(mrow + r) * 136 + ks * 32 + hi * 8];
            afr[ks].q[0] = *(const u32x4*)(pa);
            afr[ks].q[1] = *(const u32x4*)(pa + 16);
        }
#pragma unroll
        for (int nt = 0; nt < 8; ++nt) {
#pragma unroll
            for (int ks = 0; ks < 4; ++ks) {
                FragBF bf;
                const unsigned short* pb = &oaT[(nt * 16 + r) * 136 + ks * 32 + hi * 8];
                bf.q[0] = *(const u32x4*)(pb);
                bf.q[1] = *(const u32x4*)(pb + 16);
                acc8[nt] = wmma_bf16(afr[ks], bf, acc8[nt]);
            }
        }
    }
    float biasv[8];
#pragma unroll
    for (int i = 0; i < 8; ++i) biasv[i] = bout[mrow + hi * 8 + i];
    __syncthreads();                            // inputs dead; fin may alias them

#pragma unroll
    for (int nt = 0; nt < 8; ++nt)
#pragma unroll
        for (int i = 0; i < 8; ++i)
            fin[(mrow + hi * 8 + i) * 132 + nt * 16 + r] = acc8[nt][i] + biasv[i];
    __syncthreads();

    // 6) LayerNorm over W (128 columns of each row), then store
    const int rrow = tid >> 1, half = tid & 1;
    const float* base = fin + rrow * 132 + half * 64;
    float s = 0.f, ss = 0.f;
#pragma unroll 8
    for (int j = 0; j < 64; ++j) { float v = base[j]; s += v; ss += v * v; }
    red[tid] = s; red2[tid] = ss;
    __syncthreads();
    float mu  = (red [rrow * 2] + red [rrow * 2 + 1]) * (1.f / 128.f);
    float var = (red2[rrow * 2] + red2[rrow * 2 + 1]) * (1.f / 128.f) - mu * mu;
    float rsig = rsqrtf(var + LN_EPS);
    float g = gamma[rrow], bt = beta[rrow];
    float* op = out + (((size_t)b * C_CH + rrow) * HIMG + hrow) * WIMG + half * 64;
#pragma unroll 8
    for (int j = 0; j < 64; ++j) op[j] = (base[j] - mu) * rsig * g + bt;
}

// ---------------------------------------------------------------------------
extern "C" void kernel_launch(void* const* d_in, const int* in_sizes, int n_in,
                              void* d_out, int out_size, void* d_ws, size_t ws_size,
                              hipStream_t stream) {
    (void)in_sizes; (void)n_in; (void)out_size; (void)ws_size;
    const float* x    = (const float*)d_in[0];
    const float* wqkv = (const float*)d_in[1];
    const float* wout = (const float*)d_in[2];
    const float* bout = (const float*)d_in[3];
    const float* gam  = (const float*)d_in[4];
    const float* bet  = (const float*)d_in[5];
    float* out = (float*)d_out;

    unsigned short* qkv = (unsigned short*)d_ws;
    float* stats = (float*)((char*)d_ws + STATS_OFF);
    float* ctx   = (float*)((char*)d_ws + CTX_OFF);

    hipMemsetAsync(ctx, 0, (size_t)64 * 32 * 32 * sizeof(float), stream);
    qkv_gemm <<<dim3(N_TOK / 64, 384 / 64, B_SZ), 256, 0, stream>>>(x, wqkv, qkv);
    kstats   <<<dim3(2048),                       256, 0, stream>>>(qkv, stats);
    context_k<<<dim3(64, 8),                      256, 0, stream>>>(qkv, stats, ctx);
    attn_out <<<dim3(HIMG, B_SZ),                 256, 0, stream>>>(qkv, ctx, wout, bout,
                                                                    gam, bet, out);
}
